// SelfAttention_15685220565728
// MI455X (gfx1250) — matrix operand order repaired and checked by reading, not compiled, not run on hardware
//
#include <hip/hip_runtime.h>
#include <hip/hip_bf16.h>
#include <math.h>

// ---------------------------------------------------------------------------
// Types for CDNA5 WMMA (wave32): v_wmma_f32_16x16x32_bf16
// ---------------------------------------------------------------------------
typedef __bf16 bf16;
typedef __attribute__((ext_vector_type(16))) __bf16   v16bf;
typedef __attribute__((ext_vector_type(8)))  float    v8f;
typedef __attribute__((ext_vector_type(4)))  float    v4f;
typedef __attribute__((ext_vector_type(4)))  unsigned v4u;

static constexpr int Bsz = 4;
static constexpr int C   = 128;
static constexpr int N   = 4096;   // H*W = 64*64

__device__ __forceinline__ v8f wmma_bf16(v16bf a, v16bf b, v8f c) {
  // (neg_a, A, neg_b, B, c_mod, C, reuse_a, reuse_b)
  return __builtin_amdgcn_wmma_f32_16x16x32_bf16(false, a, false, b, (short)0, c,
                                                 false, false);
}

// A-operand 16x32 bf16 tile: lane L -> row (L&15), K chunks {h*8..+7, 16+h*8..+7}
__device__ __forceinline__ v16bf load_frag_a_bf16(const bf16* base, int row0,
                                                  int ldm, int k0, int lane) {
  int half = lane >> 4;
  const bf16* p = base + (size_t)(row0 + (lane & 15)) * ldm + k0 + half * 8;
  union { v16bf v; v4u u[2]; } f;
  f.u[0] = *reinterpret_cast<const v4u*>(p);
  f.u[1] = *reinterpret_cast<const v4u*>(p + 16);
  return f.v;
}

// B-operand 32x16 bf16 tile: lane L -> col (L&15), K chunks {h*8..+7, 16+h*8..+7}
__device__ __forceinline__ v16bf load_frag_b_bf16(const bf16* base, int col0,
                                                  int ldm, int k0, int lane) {
  int half = lane >> 4;
  const bf16* p = base + (size_t)(col0 + (lane & 15)) * ldm + k0 + half * 8;
  union { v16bf v; v4u u[2]; } f;
  f.u[0] = *reinterpret_cast<const v4u*>(p);
  f.u[1] = *reinterpret_cast<const v4u*>(p + 16);
  return f.v;
}

// A-operand from an f32 matrix (attn rows), converted to bf16 on the fly
__device__ __forceinline__ v16bf load_frag_a_f32(const float* base, int row0,
                                                 int ldm, int k0, int lane) {
  int half = lane >> 4;
  const float* p = base + (size_t)(row0 + (lane & 15)) * ldm + k0 + half * 8;
  v4f f0 = *reinterpret_cast<const v4f*>(p);
  v4f f1 = *reinterpret_cast<const v4f*>(p + 4);
  v4f f2 = *reinterpret_cast<const v4f*>(p + 16);
  v4f f3 = *reinterpret_cast<const v4f*>(p + 20);
  v16bf r;
#pragma unroll
  for (int i = 0; i < 4; ++i) {
    r[i]      = (__bf16)f0[i];
    r[4 + i]  = (__bf16)f1[i];
    r[8 + i]  = (__bf16)f2[i];
    r[12 + i] = (__bf16)f3[i];
  }
  return r;
}

// C/D 16x16 f32 tile store: VGPR e, lane L -> row e + (L>=16?8:0), col L&15
__device__ __forceinline__ void store_tile_f32(float* base, int ldm, int row0,
                                               int col0, v8f acc, int lane) {
  int col   = col0 + (lane & 15);
  int rbase = row0 + (lane >> 4) * 8;
#pragma unroll
  for (int e = 0; e < 8; ++e)
    base[(size_t)(rbase + e) * ldm + col] = acc[e];
}

// ---------------------------------------------------------------------------
// Kernel 1: 1x1 conv -> ReLU -> BatchNorm(eval), store bf16 in natural (C,HW)
// layout (== row-major (N,C) under the torch .view). Optionally also store a
// (C,N)-transposed copy (needed as the WMMA B-operand for attn @ g).
// 256 threads: 64 hw columns x 4 channel-groups of 32.
// ---------------------------------------------------------------------------
__global__ void convbn_kernel(const float* __restrict__ x,
                              const float* __restrict__ w,
                              const float* __restrict__ bias,
                              const float* __restrict__ gamma,
                              const float* __restrict__ beta,
                              const float* __restrict__ mean,
                              const float* __restrict__ var,
                              bf16* __restrict__ Y, bf16* __restrict__ Ytrans,
                              int writeTrans) {
  const int nb    = N / 64;
  const int b     = blockIdx.x / nb;
  const int hw    = (blockIdx.x % nb) * 64 + (threadIdx.x & 63);
  const int chb   = (threadIdx.x >> 6) * 32;
  const float* xb = x + (size_t)b * C * N;

  float acc[32];
#pragma unroll
  for (int j = 0; j < 32; ++j) acc[j] = 0.f;

  for (int cc = 0; cc < C; ++cc) {
    float xv        = xb[(size_t)cc * N + hw];
    const float* wp = w + (size_t)chb * C + cc;
#pragma unroll
    for (int j = 0; j < 32; ++j) acc[j] += wp[(size_t)j * C] * xv;
  }

#pragma unroll
  for (int j = 0; j < 32; ++j) {
    int   ch  = chb + j;
    float inv = gamma[ch] * rsqrtf(var[ch] + 1e-5f);
    float a   = acc[j] + bias[ch];
    a         = a > 0.f ? a : 0.f;                 // ReLU before BN (per _conv_bn)
    float v   = a * inv + (beta[ch] - mean[ch] * inv);
    bf16  bv  = (__bf16)v;
    size_t flat = (size_t)ch * N + hw;             // natural (C,HW) flat index
    Y[(size_t)b * C * N + flat] = bv;
    if (writeTrans) {                               // (N,C)-view transpose: (C,N)
      int n2 = (int)(flat >> 7);                    // flat / C
      int c2 = (int)(flat & 127);                   // flat % C
      Ytrans[(size_t)b * C * N + (size_t)c2 * N + n2] = bv;
    }
  }
}

// ---------------------------------------------------------------------------
// Kernel 2: raw attention scores S = theta @ phi^T  (per batch: 4096x4096, K=128)
// 8 waves/WG arranged 2x4; each wave owns a 32x32 block (4 accum tiles),
// K unrolled 4x32 -> 16 v_wmma_f32_16x16x32_bf16 per wave.
// ---------------------------------------------------------------------------
__global__ void attn_scores_kernel(const bf16* __restrict__ Theta,
                                   const bf16* __restrict__ Phi,
                                   float* __restrict__ attn) {
  const int lane = threadIdx.x & 31;
  const int wid  = threadIdx.x >> 5;
  const int b    = blockIdx.z;
  const int row0 = blockIdx.y * 64 + (wid >> 2) * 32;
  const int col0 = blockIdx.x * 128 + (wid & 3) * 32;

  const bf16* A  = Theta + (size_t)b * N * C;
  const bf16* Bm = Phi + (size_t)b * N * C;

  v8f a00 = {0.f, 0.f, 0.f, 0.f, 0.f, 0.f, 0.f, 0.f};
  v8f a01 = a00, a10 = a00, a11 = a00;

#pragma unroll
  for (int k0 = 0; k0 < C; k0 += 32) {
    v16bf fa0 = load_frag_a_bf16(A, row0, C, k0, lane);
    v16bf fa1 = load_frag_a_bf16(A, row0 + 16, C, k0, lane);
    v16bf fb0 = load_frag_b_bf16(Bm, col0, C, k0, lane);
    v16bf fb1 = load_frag_b_bf16(Bm, col0 + 16, C, k0, lane);
    a00 = wmma_bf16(fa0, fb0, a00);
    a01 = wmma_bf16(fa0, fb1, a01);
    a10 = wmma_bf16(fa1, fb0, a10);
    a11 = wmma_bf16(fa1, fb1, a11);
  }

  float* S = attn + (size_t)b * N * N;
  store_tile_f32(S, N, row0, col0, a00, lane);
  store_tile_f32(S, N, row0, col0 + 16, a01, lane);
  store_tile_f32(S, N, row0 + 16, col0, a10, lane);
  store_tile_f32(S, N, row0 + 16, col0 + 16, a11, lane);
}

// ---------------------------------------------------------------------------
// Kernel 3: in-place row softmax over 4096-wide rows (one 256-thread WG/row).
// ---------------------------------------------------------------------------
__global__ void softmax_kernel(float* __restrict__ attn) {
  __shared__ float red[256];
  float* p = attn + (size_t)blockIdx.x * N;
  const int tid = threadIdx.x;

  float r[16];
  float mx = -INFINITY;
#pragma unroll
  for (int i = 0; i < 16; ++i) {
    r[i] = p[tid + i * 256];
    mx   = fmaxf(mx, r[i]);
  }
  red[tid] = mx;
  __syncthreads();
  for (int s = 128; s > 0; s >>= 1) {
    if (tid < s) red[tid] = fmaxf(red[tid], red[tid + s]);
    __syncthreads();
  }
  mx = red[0];
  __syncthreads();

  float sum = 0.f;
#pragma unroll
  for (int i = 0; i < 16; ++i) {
    r[i] = __expf(r[i] - mx);
    sum += r[i];
  }
  red[tid] = sum;
  __syncthreads();
  for (int s = 128; s > 0; s >>= 1) {
    if (tid < s) red[tid] += red[tid + s];
    __syncthreads();
  }
  float is = 1.f / red[0];
#pragma unroll
  for (int i = 0; i < 16; ++i) p[tid + i * 256] = r[i] * is;
}

// ---------------------------------------------------------------------------
// Kernel 4: T = attn @ g  (per batch: (4096x4096)x(4096x128)).
// A rows come from f32 attn (converted to bf16 per fragment); B from the
// transposed g copy Gt (C x N) so each lane gets contiguous 16B loads.
// 8 waves/WG: 64 rows x all 128 cols; K loops over 4096 in steps of 32.
// ---------------------------------------------------------------------------
__global__ void attn_apply_kernel(const float* __restrict__ attn,
                                  const bf16* __restrict__ Gt,
                                  float* __restrict__ T) {
  const int lane = threadIdx.x & 31;
  const int wid  = threadIdx.x >> 5;
  const int b    = blockIdx.y;
  const int row0 = blockIdx.x * 64 + (wid >> 2) * 32;
  const int col0 = (wid & 3) * 32;

  const float* A  = attn + (size_t)b * N * N;
  const bf16*  Bm = Gt + (size_t)b * C * N;

  v8f a00 = {0.f, 0.f, 0.f, 0.f, 0.f, 0.f, 0.f, 0.f};
  v8f a01 = a00, a10 = a00, a11 = a00;

  for (int k0 = 0; k0 < N; k0 += 32) {
    v16bf fa0 = load_frag_a_f32(A, row0, N, k0, lane);
    v16bf fa1 = load_frag_a_f32(A, row0 + 16, N, k0, lane);
    v16bf fb0 = load_frag_b_bf16(Bm, col0, N, k0, lane);      // Gt row c, stride N
    v16bf fb1 = load_frag_b_bf16(Bm, col0 + 16, N, k0, lane);
    a00 = wmma_bf16(fa0, fb0, a00);
    a01 = wmma_bf16(fa0, fb1, a01);
    a10 = wmma_bf16(fa1, fb0, a10);
    a11 = wmma_bf16(fa1, fb1, a11);
  }

  float* Tb = T + (size_t)b * N * C;   // (N,C) row-major == raw (B,C,H,W) buffer
  store_tile_f32(Tb, C, row0, col0, a00, lane);
  store_tile_f32(Tb, C, row0, col0 + 16, a01, lane);
  store_tile_f32(Tb, C, row0 + 16, col0, a10, lane);
  store_tile_f32(Tb, C, row0 + 16, col0 + 16, a11, lane);
}

// ---------------------------------------------------------------------------
// Kernel 5: out-branch: 1x1 conv + bias -> BN -> ReLU -> x + t   (f32 out)
// T is read as (C,HW) per the raw reshape.
// ---------------------------------------------------------------------------
__global__ void outconv_kernel(const float* __restrict__ T,
                               const float* __restrict__ x,
                               const float* __restrict__ w,
                               const float* __restrict__ bias,
                               const float* __restrict__ gamma,
                               const float* __restrict__ beta,
                               const float* __restrict__ mean,
                               const float* __restrict__ var,
                               float* __restrict__ out) {
  const int nb    = N / 64;
  const int b     = blockIdx.x / nb;
  const int hw    = (blockIdx.x % nb) * 64 + (threadIdx.x & 63);
  const int chb   = (threadIdx.x >> 6) * 32;
  const float* tb = T + (size_t)b * C * N;

  float acc[32];
#pragma unroll
  for (int j = 0; j < 32; ++j) acc[j] = 0.f;

  for (int cc = 0; cc < C; ++cc) {
    float tv        = tb[(size_t)cc * N + hw];
    const float* wp = w + (size_t)chb * C + cc;
#pragma unroll
    for (int j = 0; j < 32; ++j) acc[j] += wp[(size_t)j * C] * tv;
  }

#pragma unroll
  for (int j = 0; j < 32; ++j) {
    int   ch  = chb + j;
    float inv = gamma[ch] * rsqrtf(var[ch] + 1e-5f);
    float v   = (acc[j] + bias[ch]) * inv + (beta[ch] - mean[ch] * inv);
    v         = v > 0.f ? v : 0.f;                  // ReLU AFTER BN on out branch
    size_t idx = (size_t)b * C * N + (size_t)ch * N + hw;
    out[idx]   = x[idx] + v;
  }
}

// ---------------------------------------------------------------------------
extern "C" void kernel_launch(void* const* d_in, const int* in_sizes, int n_in,
                              void* d_out, int out_size, void* d_ws,
                              size_t ws_size, hipStream_t stream) {
  const float* x = (const float*)d_in[0];
  // input order: x, then {w,b,gamma,beta,mean,var} for theta, phi, g, out
  const float* tw = (const float*)d_in[1],  *tb = (const float*)d_in[2];
  const float* tg = (const float*)d_in[3],  *tbe = (const float*)d_in[4];
  const float* tm = (const float*)d_in[5],  *tv = (const float*)d_in[6];
  const float* pw = (const float*)d_in[7],  *pb = (const float*)d_in[8];
  const float* pg = (const float*)d_in[9],  *pbe = (const float*)d_in[10];
  const float* pm = (const float*)d_in[11], *pv = (const float*)d_in[12];
  const float* gw = (const float*)d_in[13], *gb = (const float*)d_in[14];
  const float* gg = (const float*)d_in[15], *gbe = (const float*)d_in[16];
  const float* gm = (const float*)d_in[17], *gv = (const float*)d_in[18];
  const float* ow = (const float*)d_in[19], *ob = (const float*)d_in[20];
  const float* og = (const float*)d_in[21], *obe = (const float*)d_in[22];
  const float* om = (const float*)d_in[23], *ov = (const float*)d_in[24];

  const size_t BCN = (size_t)Bsz * C * N;       // 2,097,152 elements
  bf16* Yt  = (bf16*)d_ws;                       //  4 MiB
  bf16* Yp  = Yt + BCN;                          //  4 MiB
  bf16* Yg  = Yp + BCN;                          //  4 MiB
  bf16* Ygt = Yg + BCN;                          //  4 MiB  (g transposed, (C,N))
  float* T  = (float*)(Ygt + BCN);               //  8 MiB  f32 (N,C)

  float* out1 = (float*)d_out;                   // (B,C,H,W) residual output
  float* attn = out1 + BCN;                      // (B,N,N) attention output

  dim3 cblk(256), cgrd(Bsz * (N / 64));
  convbn_kernel<<<cgrd, cblk, 0, stream>>>(x, tw, tb, tg, tbe, tm, tv, Yt, Yt, 0);
  convbn_kernel<<<cgrd, cblk, 0, stream>>>(x, pw, pb, pg, pbe, pm, pv, Yp, Yp, 0);
  convbn_kernel<<<cgrd, cblk, 0, stream>>>(x, gw, gb, gg, gbe, gm, gv, Yg, Ygt, 1);

  dim3 g1(N / 128, N / 64, Bsz);                 // 32 x 64 x 4 workgroups
  attn_scores_kernel<<<g1, 256, 0, stream>>>(Yt, Yp, attn);

  softmax_kernel<<<Bsz * N, 256, 0, stream>>>(attn);

  dim3 g2(N / 64, Bsz);                          // 64 x 4 workgroups
  attn_apply_kernel<<<g2, 256, 0, stream>>>(attn, Ygt, T);

  outconv_kernel<<<cgrd, cblk, 0, stream>>>(T, x, ow, ob, og, obe, om, ov, out1);
}
